// Mamba_Encoder_Layer_67516885893471
// MI455X (gfx1250) — matrix operand order repaired and checked by reading, not compiled, not run on hardware
//
#include <hip/hip_runtime.h>
#include <hip/hip_bf16.h>

// ---------------------------------------------------------------------------
// Mamba encoder layer for gfx1250 (MI455X).
// bf16 WMMA GEMMs with async global->LDS staging (ASYNCcnt) + fused scan.
// B=2, L=4096, D_MODEL=256, D_INNER=512, D_STATE=16, DT_RANK=16, D_FF=2048.
// ---------------------------------------------------------------------------

typedef __attribute__((ext_vector_type(16))) __bf16 v16bf;
typedef __attribute__((ext_vector_type(8)))  __bf16 v8bf;
typedef __attribute__((ext_vector_type(8)))  float  v8f;

#define BATCH   2
#define SEQL    4096
#define DMODEL  256
#define DINNER  512
#define DSTATE  16
#define DTRANK  16
#define DFF     2048
#define BL      (BATCH * SEQL)          // 8192 rows in every GEMM
#define XDBLN   (DTRANK + 2 * DSTATE)   // 48

// ---------------------------------------------------------------------------
// CDNA5 async global->LDS copy (tracked by ASYNCcnt, bypasses VGPRs).
// LDS operand: low 32 bits of the generic pointer == LDS byte offset
// (ISA 10.2, LDS aperture).
// ---------------------------------------------------------------------------
__device__ __forceinline__ void async_ld_b128(unsigned ldsoff, const void* g) {
  asm volatile("global_load_async_to_lds_b128 %0, %1, off"
               :: "v"(ldsoff), "v"(g) : "memory");
}
__device__ __forceinline__ void wait_async0() {
  asm volatile("s_wait_asynccnt 0x0" ::: "memory");
}

// ---------------------------------------------------------------------------
// fp32 -> bf16 conversion (weights converted once per launch; intermediates
// are emitted in bf16 directly by GEMM/scan epilogues).
// ---------------------------------------------------------------------------
__global__ void cvt_bf16_kernel(const float* __restrict__ in,
                                __bf16* __restrict__ out, int n) {
  int i = blockIdx.x * blockDim.x + threadIdx.x;
  if (i < n) out[i] = (__bf16)in[i];
}

// ---------------------------------------------------------------------------
// Generic bf16 WMMA GEMM: C[M,N] = act(A[M,K] @ W[N,K]^T + bias)
//  - 128x128 block tile, 128 threads = 4 waves, wave tile = 64x64
//  - K-loop in 32-chunks, double-buffered LDS fed by async global->LDS copies
//  - v_wmma_f32_16x16x32_bf16, fp32 accumulation (16 WMMA per chunk per wave)
//  - fragments per CDNA5 ISA 7.12.2 16-bit A/B layouts (b128 LDS reads)
// ---------------------------------------------------------------------------
__global__ __launch_bounds__(128) void gemm_bf16_wmma(
    const __bf16* __restrict__ A,   // M x K (row-major, K contiguous)
    const __bf16* __restrict__ W,   // N x K (row-major, K contiguous)
    const float*  __restrict__ bias,// N or nullptr
    float*        __restrict__ outF,// M x N fp32 or nullptr
    __bf16*       __restrict__ outB,// M x N bf16 or nullptr
    int M, int N, int K, int relu)
{
  constexpr int TM = 128, TN = 128, TK = 32, LDT = 40;  // LDT: 80B padded row
  __shared__ __align__(16) __bf16 smem[2 * (TM + TN) * LDT];
  const unsigned smem_base = (unsigned)(size_t)(void*)smem;
  const unsigned bufBytes  = (TM + TN) * LDT * 2;       // 20480 B per buffer

  const int tid  = threadIdx.x;
  const int wave = tid >> 5;       // 0..3
  const int lane = tid & 31;
  const int hf   = lane >> 4;      // lane half (ISA A/B layout selector)
  const int l16  = lane & 15;
  const int wm   = (wave >> 1) * 64;   // wave M offset within tile (0/64)
  const int wn   = (wave & 1)  * 64;   // wave N offset within tile (0/64)
  const int tile_m = blockIdx.y * TM;
  const int tile_n = blockIdx.x * TN;
  const int r = tid;               // 0..127: row staged by this thread

  // issue one 32-wide K-chunk of A and W into LDS buffer `buf` (8 async b128)
  auto issue = [&](int k0, int buf) {
    const __bf16* as = A + (size_t)(tile_m + r) * K + k0;
    unsigned la = smem_base + (unsigned)buf * bufBytes + (unsigned)r * (LDT * 2);
    async_ld_b128(la,      as);
    async_ld_b128(la + 16, as + 8);
    async_ld_b128(la + 32, as + 16);
    async_ld_b128(la + 48, as + 24);
    int wr = tile_n + r;
    if (wr >= N) wr = N - 1;       // clamp: garbage cols are never stored
    const __bf16* ws = W + (size_t)wr * K + k0;
    unsigned lb = la + (unsigned)TM * (LDT * 2);
    async_ld_b128(lb,      ws);
    async_ld_b128(lb + 16, ws + 8);
    async_ld_b128(lb + 32, ws + 16);
    async_ld_b128(lb + 48, ws + 24);
  };

  v8f acc[4][4];
  #pragma unroll
  for (int i = 0; i < 4; ++i)
    #pragma unroll
    for (int j = 0; j < 4; ++j)
      #pragma unroll
      for (int v = 0; v < 8; ++v) acc[i][j][v] = 0.0f;

  const int niter = K / TK;
  issue(0, 0);                     // prologue: chunk 0 -> buffer 0

  for (int it = 0; it < niter; ++it) {
    const int buf = it & 1;
    wait_async0();                 // current buffer's async copies done
    __syncthreads();               // visible to all waves
    if (it + 1 < niter) issue((it + 1) * TK, buf ^ 1);  // overlap w/ compute

    const __bf16* sA = smem + buf * (TM + TN) * LDT;
    const __bf16* sB = sA + TM * LDT;

    v16bf bfrag[4];
    #pragma unroll
    for (int j = 0; j < 4; ++j) {
      // B (KxN): lane l16 = column n, K = hf*8 + [0..7] and 16 + hf*8 + [0..7]
      const __bf16* p = sB + (wn + j * 16 + l16) * LDT + hf * 8;
      v8bf lo = *(const v8bf*)p;
      v8bf hi = *(const v8bf*)(p + 16);
      bfrag[j] = __builtin_shufflevector(lo, hi,
                   0,1,2,3,4,5,6,7,8,9,10,11,12,13,14,15);
    }
    #pragma unroll
    for (int i = 0; i < 4; ++i) {
      // A (MxK): lane l16 = row m, K = hf*8 + [0..7] and 16 + hf*8 + [0..7]
      const __bf16* p = sA + (wm + i * 16 + l16) * LDT + hf * 8;
      v8bf lo = *(const v8bf*)p;
      v8bf hi = *(const v8bf*)(p + 16);
      v16bf afrag = __builtin_shufflevector(lo, hi,
                      0,1,2,3,4,5,6,7,8,9,10,11,12,13,14,15);
      #pragma unroll
      for (int j = 0; j < 4; ++j)
        acc[i][j] = __builtin_amdgcn_wmma_f32_16x16x32_bf16(
            false, afrag, false, bfrag[j],
            (short)0, acc[i][j], false, false);
    }
    __syncthreads();               // protect buffer before it is re-filled
  }

  // ---- epilogue: C/D layout -> element v is (m = hf*8 + v, n = l16) ----
  #pragma unroll
  for (int j = 0; j < 4; ++j) {
    const int gn = tile_n + wn + j * 16 + l16;
    if (gn >= N) continue;
    const float bv = bias ? bias[gn] : 0.0f;
    #pragma unroll
    for (int i = 0; i < 4; ++i) {
      #pragma unroll
      for (int v = 0; v < 8; ++v) {
        const int gm = tile_m + wm + i * 16 + hf * 8 + v;
        float val = acc[i][j][v] + bv;
        if (relu) val = fmaxf(val, 0.0f);
        const size_t o = (size_t)gm * N + gn;
        if (outF) outF[o] = val;
        if (outB) outB[o] = (__bf16)val;
      }
    }
  }
}

// ---------------------------------------------------------------------------
// Causal depthwise conv1d (k=4) + bias + SiLU on xin = xz[:, :, 0:512]
// ---------------------------------------------------------------------------
__global__ __launch_bounds__(256) void conv_silu_kernel(
    const float* __restrict__ xz, const float* __restrict__ cw,
    const float* __restrict__ cb, float* __restrict__ uF,
    __bf16* __restrict__ uB)
{
  const int idx = blockIdx.x * blockDim.x + threadIdx.x;  // BL*DINNER threads
  const int d  = idx & (DINNER - 1);
  const int bl = idx >> 9;
  const int l  = bl & (SEQL - 1);
  float acc = cb[d];
  #pragma unroll
  for (int j = 0; j < 4; ++j) {
    const int ll = l - 3 + j;
    if (ll >= 0)
      acc += xz[(size_t)(bl - 3 + j) * (2 * DINNER) + d] * cw[d * 4 + j];
  }
  const float s = acc / (1.0f + __expf(-acc));   // SiLU
  uF[idx] = s;
  uB[idx] = (__bf16)s;
}

// ---------------------------------------------------------------------------
// delta = softplus(x_dbl[:, :DTRANK] @ dt_proj_w^T + dt_proj_b)   (K=16: VALU)
// ---------------------------------------------------------------------------
__global__ __launch_bounds__(256) void dtproj_kernel(
    const float* __restrict__ xdbl, const float* __restrict__ dtw,
    const float* __restrict__ dtb, float* __restrict__ delta)
{
  const int idx = blockIdx.x * blockDim.x + threadIdx.x;  // BL*DINNER threads
  const int d   = idx & (DINNER - 1);
  const int row = idx >> 9;
  float acc = dtb[d];
  const float* xr = xdbl + (size_t)row * XDBLN;
  #pragma unroll
  for (int k = 0; k < DTRANK; ++k) acc += xr[k] * dtw[d * DTRANK + k];
  delta[idx] = (acc > 20.0f) ? acc : log1pf(__expf(acc));  // softplus
}

// ---------------------------------------------------------------------------
// Selective-scan: thread = (channel d, state n). Wave32 = 2 channels x 16
// states; y = sum_n h*C via 4 shfl_xor hops; gating fused; bf16 out for
// the out_proj GEMM.  Grid: BATCH * DINNER/16 blocks of 256 threads.
// ---------------------------------------------------------------------------
__global__ __launch_bounds__(256) void ssm_scan_kernel(
    const float* __restrict__ delta, const float* __restrict__ u,
    const float* __restrict__ xdbl,  const float* __restrict__ A_log,
    const float* __restrict__ Dp,    const float* __restrict__ xz,
    __bf16* __restrict__ yB)
{
  const int tid  = threadIdx.x;
  const int n    = tid & 15;
  const int dloc = tid >> 4;                           // 0..15
  const int b    = blockIdx.x >> 5;                    // 32 blocks per batch
  const int d    = ((blockIdx.x & 31) << 4) + dloc;

  const float Av = -__expf(A_log[d * DSTATE + n]);
  const float Dv = Dp[d];
  const float* dl = delta + (size_t)b * SEQL * DINNER + d;
  const float* uu = u     + (size_t)b * SEQL * DINNER + d;
  const float* xd = xdbl  + (size_t)b * SEQL * XDBLN;
  const float* zz = xz    + (size_t)b * SEQL * (2 * DINNER) + DINNER + d;
  __bf16*      yo = yB    + (size_t)b * SEQL * DINNER + d;

  float h = 0.0f;
  for (int t = 0; t < SEQL; ++t) {
    const float dt_v = dl[(size_t)t * DINNER];
    const float u_v  = uu[(size_t)t * DINNER];
    const float Bv   = xd[t * XDBLN + DTRANK + n];
    const float Cv   = xd[t * XDBLN + DTRANK + DSTATE + n];
    h = fmaf(__expf(dt_v * Av), h, dt_v * Bv * u_v);
    float yp = h * Cv;
    yp += __shfl_xor(yp, 1, 32);
    yp += __shfl_xor(yp, 2, 32);
    yp += __shfl_xor(yp, 4, 32);
    yp += __shfl_xor(yp, 8, 32);
    if (n == 0) {
      const float zv = zz[(size_t)t * (2 * DINNER)];
      const float yv = (yp + u_v * Dv) * (zv / (1.0f + __expf(-zv)));
      yo[(size_t)t * DINNER] = (__bf16)yv;
    }
  }
}

// ---------------------------------------------------------------------------
// Host launcher
// ---------------------------------------------------------------------------
extern "C" void kernel_launch(void* const* d_in, const int* in_sizes, int n_in,
                              void* d_out, int out_size, void* d_ws, size_t ws_size,
                              hipStream_t stream) {
  const float* x         = (const float*)d_in[0];
  const float* in_proj_w = (const float*)d_in[1];
  const float* conv_w    = (const float*)d_in[2];
  const float* conv_b    = (const float*)d_in[3];
  const float* x_proj_w  = (const float*)d_in[4];
  const float* dt_proj_w = (const float*)d_in[5];
  const float* dt_proj_b = (const float*)d_in[6];
  const float* A_log     = (const float*)d_in[7];
  const float* Dvec      = (const float*)d_in[8];
  const float* out_proj_w= (const float*)d_in[9];
  const float* lin1_w    = (const float*)d_in[10];
  const float* lin1_b    = (const float*)d_in[11];
  const float* lin2_w    = (const float*)d_in[12];
  const float* lin2_b    = (const float*)d_in[13];

  char* p = (char*)d_ws;
  auto alloc = [&](size_t bytes) -> char* {
    char* r = p;
    p += (bytes + 255) & ~(size_t)255;
    return r;
  };

  // bf16 weight copies
  __bf16* w_in  = (__bf16*)alloc((size_t)2 * DINNER * DMODEL * 2); // 1024x256
  __bf16* w_xp  = (__bf16*)alloc((size_t)XDBLN * DINNER * 2);      // 48x512
  __bf16* w_op  = (__bf16*)alloc((size_t)DMODEL * DINNER * 2);     // 256x512
  __bf16* w_l1  = (__bf16*)alloc((size_t)DFF * DMODEL * 2);        // 2048x256
  __bf16* w_l2  = (__bf16*)alloc((size_t)DFF * DFF * 2);           // 2048x2048
  // activations
  __bf16* x_bf  = (__bf16*)alloc((size_t)BL * DMODEL * 2);
  float*  xz    = (float*) alloc((size_t)BL * 2 * DINNER * 4);
  float*  uF    = (float*) alloc((size_t)BL * DINNER * 4);
  __bf16* uB    = (__bf16*)alloc((size_t)BL * DINNER * 2);
  float*  xdbl  = (float*) alloc((size_t)BL * XDBLN * 4);
  float*  delta = (float*) alloc((size_t)BL * DINNER * 4);
  __bf16* y_bf  = (__bf16*)alloc((size_t)BL * DINNER * 2);
  __bf16* m_bf  = (__bf16*)alloc((size_t)BL * DMODEL * 2);
  __bf16* h1_bf = (__bf16*)alloc((size_t)BL * DFF * 2);

  auto cvt = [&](const float* src, __bf16* dst, int n) {
    cvt_bf16_kernel<<<(n + 255) / 256, 256, 0, stream>>>(src, dst, n);
  };
  cvt(x,          x_bf, BL * DMODEL);
  cvt(in_proj_w,  w_in, 2 * DINNER * DMODEL);
  cvt(x_proj_w,   w_xp, XDBLN * DINNER);
  cvt(out_proj_w, w_op, DMODEL * DINNER);
  cvt(lin1_w,     w_l1, DFF * DMODEL);
  cvt(lin2_w,     w_l2, DFF * DFF);

  const int MT = BL / 128;  // 64 M-tiles

  // 1) in_proj: (8192,256) @ (1024,256)^T -> xz fp32
  gemm_bf16_wmma<<<dim3((2 * DINNER) / 128, MT), 128, 0, stream>>>(
      x_bf, w_in, nullptr, xz, nullptr, BL, 2 * DINNER, DMODEL, 0);

  // 2) causal depthwise conv + SiLU -> u (fp32 + bf16)
  conv_silu_kernel<<<(BL * DINNER) / 256, 256, 0, stream>>>(
      xz, conv_w, conv_b, uF, uB);

  // 3) x_proj: (8192,512) @ (48,512)^T -> x_dbl fp32
  gemm_bf16_wmma<<<dim3((XDBLN + 127) / 128, MT), 128, 0, stream>>>(
      uB, w_xp, nullptr, xdbl, nullptr, BL, XDBLN, DINNER, 0);

  // 4) dt_proj + softplus -> delta
  dtproj_kernel<<<(BL * DINNER) / 256, 256, 0, stream>>>(
      xdbl, dt_proj_w, dt_proj_b, delta);

  // 5) selective scan + D-skip + SiLU gating -> y (bf16)
  ssm_scan_kernel<<<BATCH * (DINNER / 16), 256, 0, stream>>>(
      delta, uF, xdbl, A_log, Dvec, xz, y_bf);

  // 6) out_proj: (8192,512) @ (256,512)^T -> m (bf16)
  gemm_bf16_wmma<<<dim3(DMODEL / 128, MT), 128, 0, stream>>>(
      y_bf, w_op, nullptr, nullptr, m_bf, BL, DMODEL, DINNER, 0);

  // 7) lin1 + ReLU: (8192,256) @ (2048,256)^T -> h1 (bf16)
  gemm_bf16_wmma<<<dim3(DFF / 128, MT), 128, 0, stream>>>(
      m_bf, w_l1, lin1_b, nullptr, h1_bf, BL, DFF, DMODEL, 1);

  // 8) lin2: (8192,2048) @ (2048,2048)^T -> d_out fp32
  gemm_bf16_wmma<<<dim3(DFF / 128, MT), 128, 0, stream>>>(
      h1_bf, w_l2, lin2_b, (float*)d_out, nullptr, BL, DFF, DFF, 0);
}
